// MultiHeadAttention_80779744903181
// MI455X (gfx1250) — matrix operand order repaired and checked by reading, not compiled, not run on hardware
//
#include <hip/hip_runtime.h>
#include <hip/hip_bf16.h>

#define T_SEQ   4096
#define D_EMBD  1024
#define N_HEADS 16
#define D_HEAD  64
#define D3      (3 * D_EMBD)

typedef __attribute__((ext_vector_type(16))) __bf16 v16bf;
typedef __attribute__((ext_vector_type(8)))  __bf16 v8bf;
typedef __attribute__((ext_vector_type(8)))  float  v8f;

static __device__ __forceinline__ v16bf combine8(v8bf lo, v8bf hi) {
    return __builtin_shufflevector(lo, hi, 0, 1, 2, 3, 4, 5, 6, 7,
                                            8, 9, 10, 11, 12, 13, 14, 15);
}

// A-fragment (16x32, row-major source, stride lda):
// lane: m = row, half = lane>>4.  element i: K = (i>>3)*16 + half*8 + (i&7)
static __device__ __forceinline__ v16bf load_a_frag(const __bf16* __restrict__ A,
                                                    size_t row, int lda, int k0, int half) {
    const __bf16* p = A + row * (size_t)lda + k0 + half * 8;
    return combine8(*(const v8bf*)p, *(const v8bf*)(p + 16));
}

// B-fragment (32x16, source stored N-major: Bt[n][k], stride ldb):
// lane: n = col, half = lane>>4.  element i: K = (i>>3)*16 + half*8 + (i&7)
static __device__ __forceinline__ v16bf load_b_frag(const __bf16* __restrict__ Bt,
                                                    size_t col, int ldb, int k0, int half) {
    const __bf16* p = Bt + col * (size_t)ldb + k0 + half * 8;
    return combine8(*(const v8bf*)p, *(const v8bf*)(p + 16));
}

static __device__ __forceinline__ v8f wmma_bf16(v16bf a, v16bf b, v8f c) {
    return __builtin_amdgcn_wmma_f32_16x16x32_bf16(false, a, false, b,
                                                   (short)0, c, false, false);
}

// ---------------------------------------------------------------- converters
__global__ void cast_bf16_kernel(const float* __restrict__ in,
                                 __bf16* __restrict__ out, int n) {
    int i = blockIdx.x * blockDim.x + threadIdx.x;
    if (i < n) out[i] = (__bf16)in[i];
}

// W is K x N (row-major); Wt is N x K (row-major) bf16
__global__ void transpose_cast_kernel(const float* __restrict__ W,
                                      __bf16* __restrict__ Wt, int K, int N) {
    int i = blockIdx.x * blockDim.x + threadIdx.x;
    if (i >= K * N) return;
    int n = i / K;
    int k = i - n * K;
    Wt[(size_t)n * K + k] = (__bf16)W[(size_t)k * N + n];
}

// ---------------------------------------------------------------- GEMM
// C[M x N] = A_bf16[M x K] @ Bt_bf16[N x K]^T + bias[N]
// block: 256 threads = 8 waves, block tile 128x128, wave tile 32x64
template <bool OUT_BF16>
__global__ void gemm_bf16_kernel(const __bf16* __restrict__ A,
                                 const __bf16* __restrict__ Bt,
                                 const float* __restrict__ bias,
                                 void* __restrict__ C,
                                 int M, int N, int K) {
    const int lane  = threadIdx.x & 31;
    const int wave  = threadIdx.x >> 5;
    const int waveM = wave >> 1;           // 0..3
    const int waveN = wave & 1;            // 0..1
    const int m0    = blockIdx.x * 128 + waveM * 32;
    const int n0    = blockIdx.y * 128 + waveN * 64;
    const int half  = lane >> 4;
    const int l16   = lane & 15;

    v8f acc[2][4] = {};

    for (int k0 = 0; k0 < K; k0 += 32) {
        v16bf a[2], b[4];
#pragma unroll
        for (int i = 0; i < 2; ++i) {
            const __bf16* p = A + (size_t)(m0 + i * 16 + l16) * K + k0 + half * 8;
            a[i] = combine8(*(const v8bf*)p, *(const v8bf*)(p + 16));
            __builtin_prefetch(p + 32, 0, 3);     // next k-tile -> global_prefetch_b8
        }
#pragma unroll
        for (int j = 0; j < 4; ++j) {
            const __bf16* p = Bt + (size_t)(n0 + j * 16 + l16) * K + k0 + half * 8;
            b[j] = combine8(*(const v8bf*)p, *(const v8bf*)(p + 16));
            __builtin_prefetch(p + 32, 0, 3);
        }
#pragma unroll
        for (int i = 0; i < 2; ++i)
#pragma unroll
            for (int j = 0; j < 4; ++j)
                acc[i][j] = wmma_bf16(a[i], b[j], acc[i][j]);
    }

#pragma unroll
    for (int j = 0; j < 4; ++j) {
        const int n  = n0 + j * 16 + l16;
        const float bv = bias[n];
#pragma unroll
        for (int i = 0; i < 2; ++i) {
#pragma unroll
            for (int r = 0; r < 8; ++r) {
                const int m = m0 + i * 16 + r + half * 8;
                const float v = acc[i][j][r] + bv;
                if (OUT_BF16)
                    ((__bf16*)C)[(size_t)m * N + n] = (__bf16)v;
                else
                    ((float*)C)[(size_t)m * N + n] = v;
            }
        }
    }
}

// ---------------------------------------------------------------- split heads
// kqv bf16 [T][3*D]; columns: [0,D)=k, [D,2D)=q, [2D,3D)=v
// Qh,Kh: [H][T][64] row-major;  Vt: [H][64][T] (d-major, ready as B-fragment)
__global__ void split_heads_kernel(const __bf16* __restrict__ kqv,
                                   __bf16* __restrict__ Qh,
                                   __bf16* __restrict__ Kh,
                                   __bf16* __restrict__ Vt) {
    int i = blockIdx.x * blockDim.x + threadIdx.x;
    if (i >= T_SEQ * D_EMBD) return;
    int t = i >> 10;
    int c = i & (D_EMBD - 1);
    int h = c >> 6;
    int d = c & (D_HEAD - 1);
    const size_t row = (size_t)t * D3;
    Kh[((size_t)h * T_SEQ + t) * D_HEAD + d] = kqv[row + c];
    Qh[((size_t)h * T_SEQ + t) * D_HEAD + d] = kqv[row + D_EMBD + c];
    Vt[((size_t)h * D_HEAD + d) * T_SEQ + t] = kqv[row + 2 * D_EMBD + c];
}

// ---------------------------------------------------------------- attention
// One 32-wide s-chunk for TWO 16-row tiles sharing the same K/V fragments:
//   S[t] = tril(Q[t] K^T)  (4 WMMAs per tile), bounce S through LDS
//   (D-frag -> A-frag relayout), then O[t] += S[t] @ V (4 WMMAs per tile).
// K fragments: 128 B/lane, V fragments: 128 B/lane, shared by both tiles
// -> 16 WMMAs (262 kFLOP/wave) per 8 KB/wave of L2 traffic (32 flops/B).
// MASK instantiated only for the single chunk at s0 == r0 (r0 % 32 == 0):
// it is the diagonal chunk for tile0 and the partial chunk for tile1.
template <bool MASK>
static __device__ __forceinline__ void attn_chunk(const __bf16* __restrict__ Kp,
                                                  const __bf16* __restrict__ Vp,
                                                  __bf16* __restrict__ sl,
                                                  const v16bf aq[2][2], v8f o[2][4],
                                                  int s0, int r0, int half, int l16) {
#pragma unroll
    for (int sub = 0; sub < 2; ++sub) {
        const int ss = s0 + sub * 16;
        v16bf bk0 = load_b_frag(Kp, (size_t)(ss + l16), D_HEAD, 0, half);
        v16bf bk1 = load_b_frag(Kp, (size_t)(ss + l16), D_HEAD, 32, half);
#pragma unroll
        for (int t = 0; t < 2; ++t) {
            v8f s = {};
            s = wmma_bf16(aq[t][0], bk0, s);
            s = wmma_bf16(aq[t][1], bk1, s);
#pragma unroll
            for (int r = 0; r < 8; ++r) {
                float v = s[r];
                if (MASK) {
                    const int trow = r0 + t * 16 + r + half * 8;
                    const int sg   = ss + l16;
                    v = (sg <= trow) ? v : 0.0f;
                }
                // tile t staged at element offset t*512, row stride 32
                sl[t * 512 + (r + half * 8) * 32 + sub * 16 + l16] = (__bf16)v;
            }
        }
    }
    // same-wave DS ops are in-order; wait before consuming loaded values
    asm volatile("s_wait_dscnt 0" ::: "memory");

    v16bf as[2];
#pragma unroll
    for (int t = 0; t < 2; ++t) {
        const __bf16* p = sl + t * 512 + l16 * 32 + half * 8;
        as[t] = combine8(*(const v8bf*)p, *(const v8bf*)(p + 16));
    }
#pragma unroll
    for (int n = 0; n < 4; ++n) {
        v16bf bv = load_b_frag(Vp, (size_t)(n * 16 + l16), T_SEQ, s0, half);
#pragma unroll
        for (int t = 0; t < 2; ++t)
            o[t][n] = wmma_bf16(as[t], bv, o[t][n]);
    }
    asm volatile("" ::: "memory");
}

// block = (row-tile of 128, head); 4 waves, each wave: 32 rows (2 tiles), d=64.
// __launch_bounds__(128, 1): occupancy floor 1 wave/SIMD so the full fragment
// working set (~220 VGPRs) stays in registers -- no scratch spills.
// O_bf16: [T][D] with head h occupying columns [h*64, h*64+64)
__global__ void __launch_bounds__(128, 1)
attention_kernel(const __bf16* __restrict__ Qh,
                 const __bf16* __restrict__ Kh,
                 const __bf16* __restrict__ Vt,
                 __bf16* __restrict__ O) {
    __shared__ __bf16 lds[4][2 * 16 * 32];   // per-wave 2x(16x32) bf16 S staging

    const int lane = threadIdx.x & 31;
    const int wave = threadIdx.x >> 5;       // 0..3
    const int h    = blockIdx.y;
    const int r0   = blockIdx.x * 128 + wave * 32;   // multiple of 32
    const int half = lane >> 4;
    const int l16  = lane & 15;

    const __bf16* Q  = Qh + (size_t)h * T_SEQ * D_HEAD;
    const __bf16* Kp = Kh + (size_t)h * T_SEQ * D_HEAD;
    const __bf16* Vp = Vt + (size_t)h * D_HEAD * T_SEQ;

    // Q fragments: 2 row-tiles x K-dim halves {0..31},{32..63}
    v16bf aq[2][2];
#pragma unroll
    for (int t = 0; t < 2; ++t) {
        aq[t][0] = load_a_frag(Q, (size_t)(r0 + t * 16 + l16), D_HEAD, 0, half);
        aq[t][1] = load_a_frag(Q, (size_t)(r0 + t * 16 + l16), D_HEAD, 32, half);
    }

    v8f o[2][4] = {};
    __bf16* sl = &lds[wave][0];

    // fully-unmasked chunks for both tiles: s0 + 31 < r0
    const int nFull = r0 >> 5;
    for (int c = 0; c < nFull; ++c)
        attn_chunk<false>(Kp, Vp, sl, aq, o, c * 32, r0, half, l16);
    // single masked chunk at s0 == r0 finishes the causal range of both tiles
    attn_chunk<true>(Kp, Vp, sl, aq, o, r0, r0, half, l16);

    // write O tiles (bf16, row-major T x D)
#pragma unroll
    for (int t = 0; t < 2; ++t) {
#pragma unroll
        for (int n = 0; n < 4; ++n) {
#pragma unroll
            for (int r = 0; r < 8; ++r) {
                const int trow = r0 + t * 16 + r + half * 8;
                O[(size_t)trow * D_EMBD + h * D_HEAD + n * 16 + l16] = (__bf16)o[t][n][r];
            }
        }
    }
}

// ---------------------------------------------------------------- launcher
extern "C" void kernel_launch(void* const* d_in, const int* in_sizes, int n_in,
                              void* d_out, int out_size, void* d_ws, size_t ws_size,
                              hipStream_t stream) {
    const float* x  = (const float*)d_in[0];   // (T, D)
    const float* W1 = (const float*)d_in[1];   // (D, 3D)
    const float* b1 = (const float*)d_in[2];   // (3D,)
    const float* W2 = (const float*)d_in[3];   // (D, D)
    const float* b2 = (const float*)d_in[4];   // (D,)
    float* out = (float*)d_out;                // (T, D)

    char* ws = (char*)d_ws;
    size_t off = 0;
    auto alloc = [&](size_t bytes) {
        char* p = ws + off;
        off += (bytes + 255) & ~(size_t)255;
        return p;
    };
    __bf16* xb   = (__bf16*)alloc((size_t)T_SEQ * D_EMBD * 2);   //  8 MB
    __bf16* W1t  = (__bf16*)alloc((size_t)D3 * D_EMBD * 2);      //  6 MB
    __bf16* W2t  = (__bf16*)alloc((size_t)D_EMBD * D_EMBD * 2);  //  2 MB
    __bf16* kqv  = (__bf16*)alloc((size_t)T_SEQ * D3 * 2);       // 24 MB
    __bf16* Qh   = (__bf16*)alloc((size_t)T_SEQ * D_EMBD * 2);   //  8 MB
    __bf16* Kh   = (__bf16*)alloc((size_t)T_SEQ * D_EMBD * 2);   //  8 MB
    __bf16* Vt   = (__bf16*)alloc((size_t)T_SEQ * D_EMBD * 2);   //  8 MB
    __bf16* Obuf = (__bf16*)alloc((size_t)T_SEQ * D_EMBD * 2);   //  8 MB
    (void)ws_size;

    const int thr = 256;

    // 1) casts / transposes into bf16
    {
        int n = T_SEQ * D_EMBD;
        cast_bf16_kernel<<<(n + thr - 1) / thr, thr, 0, stream>>>(x, xb, n);
    }
    {
        int n = D_EMBD * D3;
        transpose_cast_kernel<<<(n + thr - 1) / thr, thr, 0, stream>>>(W1, W1t, D_EMBD, D3);
    }
    {
        int n = D_EMBD * D_EMBD;
        transpose_cast_kernel<<<(n + thr - 1) / thr, thr, 0, stream>>>(W2, W2t, D_EMBD, D_EMBD);
    }

    // 2) kqv = x @ W1 + b1   (bf16 out)
    gemm_bf16_kernel<true><<<dim3(T_SEQ / 128, D3 / 128), thr, 0, stream>>>(
        xb, W1t, b1, kqv, T_SEQ, D3, D_EMBD);

    // 3) split into head-major Q, K and d-major V
    {
        int n = T_SEQ * D_EMBD;
        split_heads_kernel<<<(n + thr - 1) / thr, thr, 0, stream>>>(kqv, Qh, Kh, Vt);
    }

    // 4) causal retention: O = tril(Q K^T) V
    attention_kernel<<<dim3(T_SEQ / 128, N_HEADS), 128, 0, stream>>>(Qh, Kh, Vt, Obuf);

    // 5) out = O @ W2 + b2   (f32 out)
    gemm_bf16_kernel<false><<<dim3(T_SEQ / 128, D_EMBD / 128), thr, 0, stream>>>(
        Obuf, W2t, b2, out, T_SEQ, D_EMBD, D_EMBD);
}
